// ContextAwareModel_25374666785570
// MI455X (gfx1250) — matrix operand order repaired and checked by reading, not compiled, not run on hardware
//
#include <hip/hip_runtime.h>
#include <math.h>

// ---------------------------------------------------------------------------
// ContextAwareModel: bidirectional(-weights) LSTM scan, T=512 steps.
// Per-step fused kernel using v_wmma_f32_16x16x32_f16 (f16 in, f32 acc):
//   gates[B=256, 2dirs*4H=2048] = [emb_x | h_prev] (K=1024) x W^T
// CDNA5 trick: for row-major 16-bit data the WMMA A-fragment is directly
// loadable with plain GLOBAL_LOAD_B128 (lane l = row l%16, two contiguous
// 16B chunks at K-offsets (l/16)*8 and 16+(l/16)*8). Embedding table is
// pre-converted to f16 once (12.3MB, L2-resident) so BOTH A sources stream
// straight from global -> zero LDS staging, zero barriers in the K loop.
// Weights pre-swizzled once into B-fragment order (4MB, L2-resident).
// Each wave computes TWO 16x16 N-tiles (2 WMMA per A fragment); each block
// owns all 4 gate tiles of one (dir, 16-batch, 32-hidden) block, so the LSTM
// nonlinearity fuses in-kernel through a single LDS gate exchange.
// ---------------------------------------------------------------------------

typedef _Float16 v16h __attribute__((ext_vector_type(16)));
typedef _Float16 v8h  __attribute__((ext_vector_type(8)));
typedef _Float16 v4h  __attribute__((ext_vector_type(4)));
typedef float    v8f  __attribute__((ext_vector_type(8)));

#define HH   256   // hidden
#define EE   768   // embedding dim
#define BB   256   // batch
#define TT   512   // seq len
#define NKT  32    // (E+H)/32 k-tiles
#define NKE  24    // E/32 embedding k-tiles
#define NTPD 64    // n-tiles per direction (1024/16)

__device__ __forceinline__ v8f wmma_f16(const v16h a, const v16h b, v8f c) {
  return __builtin_amdgcn_wmma_f32_16x16x32_f16(
      /*neg_a=*/false, a, /*neg_b=*/false, b,
      /*c_mod=*/(short)0, c, /*reuse_a=*/false, /*reuse_b=*/false);
}

// ---------------------------------------------------------------------------
// Pre-swizzle f32 weights into f16 WMMA B-fragment order.
// B-fragment (32x16 f16): lane l -> N = l%16; element j -> K = ktile*32 + (l/16)*8 + (j&7) + (j>>3)*16.
// Flat: wsW[((d*64 + ntile)*32 + ktile)*32 + lane][16 halves]
// Combined K: k<768 -> W_ih[n][k], else W_hh[n][k-768].
// ---------------------------------------------------------------------------
__global__ __launch_bounds__(256)
void prep_weights(const float* __restrict__ w_ih_f, const float* __restrict__ w_hh_f,
                  const float* __restrict__ b_ih_f, const float* __restrict__ b_hh_f,
                  const float* __restrict__ w_ih_b, const float* __restrict__ w_hh_b,
                  const float* __restrict__ b_ih_b, const float* __restrict__ b_hh_b,
                  _Float16* __restrict__ wsW, float* __restrict__ biasC) {
  const int tid  = blockIdx.x * blockDim.x + threadIdx.x;   // 524288 threads
  const int base = tid * 4;                                 // 4 halves each
  const int j0    = base & 15;
  const int lane  = (base >> 4) & 31;
  const int ktile = (base >> 9) & 31;
  const int ntile = (base >> 14) & 63;
  const int d     = base >> 20;                             // 0 or 1

  const float* wih = d ? w_ih_b : w_ih_f;
  const float* whh = d ? w_hh_b : w_hh_f;
  const int n     = ntile * 16 + (lane & 15);
  const int kbase = ktile * 32 + ((lane >> 4) << 3) + (j0 & 7) + ((j0 >> 3) << 4);

  _Float16* dst = wsW + (size_t)base;
#pragma unroll
  for (int q = 0; q < 4; ++q) {
    const int k = kbase + q;
    const float v = (k < EE) ? wih[(size_t)n * EE + k]
                             : whh[(size_t)n * HH + (k - EE)];
    dst[q] = (_Float16)v;
  }
  // combined bias b_ih + b_hh, per dir: biasC[d*1024 + n]
  if (tid < 2 * 4 * HH) {
    const int dd = tid >> 10, nn = tid & 1023;
    biasC[tid] = dd ? (b_ih_b[nn] + b_hh_b[nn]) : (b_ih_f[nn] + b_hh_f[nn]);
  }
}

// one-time f32 -> f16 conversion of the embedding table (row-major preserved)
__global__ __launch_bounds__(256)
void prep_emb(const float* __restrict__ emb, _Float16* __restrict__ embh,
              const int total4) {
  const int i = blockIdx.x * blockDim.x + threadIdx.x;
  if (i >= total4) return;
  const float4 f = ((const float4*)emb)[i];
  v4h v;
  v.x = (_Float16)f.x; v.y = (_Float16)f.y;
  v.z = (_Float16)f.z; v.w = (_Float16)f.w;
  ((v4h*)embh)[i] = v;
}

// zero h0 (f16) and c (f32): 2 dirs * 256 b * 256 j = 131072 each
__global__ __launch_bounds__(256)
void init_state(_Float16* __restrict__ h0, float* __restrict__ c) {
  const int i = blockIdx.x * blockDim.x + threadIdx.x;
  h0[i] = (_Float16)0.f;
  c[i]  = 0.f;
}

__device__ __forceinline__ float sigf(float x) { return 1.f / (1.f + __expf(-x)); }

// ---------------------------------------------------------------------------
// One recurrence step.
// grid = (16 Mtiles, 16); blockIdx.y = dir*8 + jp (jp = 32-col hidden group)
// block = 128 (4 waves); wave w = gate w, two n-tiles: w*16 + jp*2 {+1}
// K loop: it<24 -> f16 embedding rows (token-gathered), it>=24 -> h_prev.
// A fragments load directly from row-major f16 global memory; no LDS, no
// barriers until the single gate-exchange sync.
// ---------------------------------------------------------------------------
__global__ __launch_bounds__(128)
void lstm_step(const int t,
               const int*   __restrict__ input_ids,   // [B, T]
               const int*   __restrict__ target_idx,  // [B]
               const _Float16* __restrict__ embh,     // [V, E] f16
               const _Float16* __restrict__ wsW,      // swizzled weights
               const float* __restrict__ biasC,       // [2][1024]
               const _Float16* __restrict__ h_in,     // [2][B][H]
               _Float16* __restrict__ h_out,          // [2][B][H]
               float* __restrict__ cbuf,              // [2][B][H]
               float* __restrict__ gathered)          // [B][2H]
{
  __shared__ float gTile[4][16][32];   // i/f/g/o exchange (16 rows x 32 cols)

  const int tid   = threadIdx.x;
  const int lane  = tid & 31;
  const int wv    = tid >> 5;           // gate index 0..3
  const int mtile = blockIdx.x;
  const int d     = blockIdx.y >> 3;
  const int jp    = blockIdx.y & 7;     // 32-wide hidden-col group
  const int mbase = mtile * 16;

  // per-lane A-row base: row M = lane%16, half-chunk select = lane/16
  const int arow  = mbase + (lane & 15);
  const int koff  = (lane >> 4) << 3;   // 0 or 8 halves
  const int tok   = input_ids[(size_t)arow * TT + t];
  const _Float16* const ae = embh + (size_t)tok * EE + koff;
  const _Float16* const ah = h_in + ((size_t)d * BB + arow) * HH + koff;

  // B fragments: two n-tiles per wave, contiguous 32B/lane loads
  const int nt0 = wv * 16 + jp * 2;
  const _Float16* const wb0 =
      wsW + ((((size_t)d * NTPD + nt0) * NKT) * 32 + lane) * 16;
  const _Float16* const wb1 = wb0 + (size_t)NKT * 512;  // next n-tile

  // A fragment = two contiguous 16B chunks of the row (K and K+16 blocks)
  auto lda = [&](const _Float16* p) -> v16h {
    union { v16h v; v8h h[2]; } u;
    u.h[0] = *(const v8h*)(p);
    u.h[1] = *(const v8h*)(p + 16);
    return u.v;
  };

  v8f acc0 = {}, acc1 = {};
#pragma unroll 4
  for (int it = 0; it < NKE; ++it) {            // embedding K-range
    const v16h a  = lda(ae + it * 32);
    const v16h b0 = *(const v16h*)(wb0 + (size_t)it * 512);
    const v16h b1 = *(const v16h*)(wb1 + (size_t)it * 512);
    acc0 = wmma_f16(a, b0, acc0);
    acc1 = wmma_f16(a, b1, acc1);
  }
#pragma unroll
  for (int it = NKE; it < NKT; ++it) {          // hidden-state K-range
    const v16h a  = lda(ah + (it - NKE) * 32);
    const v16h b0 = *(const v16h*)(wb0 + (size_t)it * 512);
    const v16h b1 = *(const v16h*)(wb1 + (size_t)it * 512);
    acc0 = wmma_f16(a, b0, acc0);
    acc1 = wmma_f16(a, b1, acc1);
  }

  // --- bias + gate exchange (C/D layout: vgpr v, lanes<16 -> M=v, else M=v+8)
  const int ncol = jp * 32 + (lane & 15);       // col of acc0 within gate
  const float bia0 = biasC[d * 1024 + wv * 256 + ncol];
  const float bia1 = biasC[d * 1024 + wv * 256 + ncol + 16];
#pragma unroll
  for (int v = 0; v < 8; ++v) {
    const int r = v + ((lane >> 4) << 3);
    gTile[wv][r][(lane & 15)]      = acc0[v] + bia0;
    gTile[wv][r][(lane & 15) + 16] = acc1[v] + bia1;
  }
  __syncthreads();

  // --- fused LSTM elementwise update (128 threads x 4 cells)
#pragma unroll
  for (int e = tid; e < 512; e += 128) {
    const int r  = e >> 5, jj = e & 31;
    const int b  = mbase + r;
    const int j  = jp * 32 + jj;
    const float gi = gTile[0][r][jj];
    const float gf = gTile[1][r][jj];
    const float gg = gTile[2][r][jj];
    const float go = gTile[3][r][jj];
    const size_t cix = ((size_t)d * BB + b) * HH + j;
    const float cn = sigf(gf) * cbuf[cix] + sigf(gi) * tanhf(gg);
    const float hn = sigf(go) * tanhf(cn);
    cbuf[cix]  = cn;
    h_out[cix] = (_Float16)hn;
    if (target_idx[b] == t) gathered[(size_t)b * (2 * HH) + d * HH + j] = hn;
  }
}

// out[b] = sigmoid( gathered[b,:] . w_out + b_out )
__global__ __launch_bounds__(64)
void finalize(const float* __restrict__ gathered,
              const float* __restrict__ w_out,
              const float* __restrict__ b_out,
              float* __restrict__ out) {
  const int b = blockIdx.x * blockDim.x + threadIdx.x;
  if (b >= BB) return;
  const float* g = gathered + (size_t)b * (2 * HH);
  float acc = 0.f;
#pragma unroll 8
  for (int i = 0; i < 2 * HH; ++i) acc = fmaf(g[i], w_out[i], acc);
  out[b] = sigf(acc + b_out[0]);
}

// ---------------------------------------------------------------------------
extern "C" void kernel_launch(void* const* d_in, const int* in_sizes, int n_in,
                              void* d_out, int out_size, void* d_ws, size_t ws_size,
                              hipStream_t stream) {
  (void)n_in; (void)out_size; (void)ws_size;
  const int*   input_ids = (const int*)  d_in[0];
  const int*   tgt       = (const int*)  d_in[1];
  const float* emb       = (const float*)d_in[3];
  const float* w_ih_f    = (const float*)d_in[4];
  const float* w_hh_f    = (const float*)d_in[5];
  const float* b_ih_f    = (const float*)d_in[6];
  const float* b_hh_f    = (const float*)d_in[7];
  const float* w_ih_b    = (const float*)d_in[8];
  const float* w_hh_b    = (const float*)d_in[9];
  const float* b_ih_b    = (const float*)d_in[10];
  const float* b_hh_b    = (const float*)d_in[11];
  const float* w_out     = (const float*)d_in[12];
  const float* b_out     = (const float*)d_in[13];
  const int    embElems  = in_sizes[3];          // V * E

  // scratch carve-up (~17.6 MB total)
  char* ws = (char*)d_ws;
  _Float16* wsW     = (_Float16*)(ws);                           // 4 MB swizzled weights
  size_t off        = (size_t)4 << 20;
  _Float16* embh    = (_Float16*)(ws + off);                     // 12.3 MB f16 embedding
  off += ((size_t)embElems * 2 + 255) & ~(size_t)255;
  float*    biasC   = (float*)(ws + off);   off += 8192;         // 8 KB combined bias
  float*    cbuf    = (float*)(ws + off);   off += (size_t)2 * BB * HH * 4;  // 512 KB
  _Float16* hA      = (_Float16*)(ws + off); off += (size_t)2 * BB * HH * 2; // 256 KB
  _Float16* hB      = (_Float16*)(ws + off); off += (size_t)2 * BB * HH * 2; // 256 KB
  float*    gathered= (float*)(ws + off);                        // 512 KB

  prep_weights<<<2048, 256, 0, stream>>>(w_ih_f, w_hh_f, b_ih_f, b_hh_f,
                                         w_ih_b, w_hh_b, b_ih_b, b_hh_b,
                                         wsW, biasC);
  const int total4 = embElems / 4;
  prep_emb<<<(total4 + 255) / 256, 256, 0, stream>>>(emb, embh, total4);
  init_state<<<512, 256, 0, stream>>>(hA, cbuf);

  const dim3 grid(16, 16);  // 16 M-tiles x (2 dirs * 8 hidden 32-col groups)
  for (int t = 0; t < TT; ++t) {
    _Float16* hin  = (t & 1) ? hB : hA;
    _Float16* hout = (t & 1) ? hA : hB;
    lstm_step<<<grid, 128, 0, stream>>>(t, input_ids, tgt, embh, wsW, biasC,
                                        hin, hout, cbuf, gathered);
  }
  finalize<<<4, 64, 0, stream>>>(gathered, w_out, b_out, (float*)d_out);
}
